// MultiHeadAttention_69466801045770
// MI455X (gfx1250) — matrix operand order repaired and checked by reading, not compiled, not run on hardware
//
#include <hip/hip_runtime.h>

typedef __bf16 bf16;
typedef __attribute__((ext_vector_type(16))) __bf16 v16bf;
typedef __attribute__((ext_vector_type(8)))  __bf16 v8bf;
typedef __attribute__((ext_vector_type(8)))  float  v8f;

#define B_  4
#define T_  2048
#define D1_ 1024
#define D2_ 768
#define H_  16
#define DT_ 64

__device__ __forceinline__ v8f wmma_bf16(v16bf a, v16bf b, v8f c) {
  // (neg_a, A, neg_b, B, c_mod, C, reuse_a, reuse_b)
  return __builtin_amdgcn_wmma_f32_16x16x32_bf16(false, a, false, b, (short)0, c, false, false);
}

__device__ __forceinline__ v8f vzero8() {
  v8f z;
#pragma unroll
  for (int i = 0; i < 8; i++) z[i] = 0.f;
  return z;
}

// ---------------------------------------------------------------------------
// Tiled GEMM: C[M,N] = A[M,K] @ W[K,N] + bias,  bf16 WMMA, f32 accumulate.
// A is f32 (A_IS_BF16==0) or bf16 (==1); W, bias, C are f32.
// Block: 256 threads = 8 waves; block tile 64x128; wave tile 16x64 (4 accums).
// ---------------------------------------------------------------------------
template <int A_IS_BF16>
__global__ void __launch_bounds__(256)
gemm_kernel(const void* __restrict__ Av, const float* __restrict__ W,
            const float* __restrict__ bias, float* __restrict__ C,
            int M, int N, int K) {
  const int n0   = blockIdx.x * 128;
  const int m0   = blockIdx.y * 64;
  const int tid  = threadIdx.x;
  const int lane = tid & 31;
  const int w    = tid >> 5;
  const int ln   = lane & 15;
  const int hl   = lane >> 4;
  const int wm   = w & 3;   // M subtile (16 rows)
  const int wn   = w >> 2;  // N subtile (64 cols -> 4 accums)

  __shared__ __align__(16) bf16 As[64 * 48];   // [row][k], stride 48 halves (96B)
  __shared__ __align__(16) bf16 Ws[128 * 48];  // [n][k] (transposed), stride 48

  v8f acc0 = vzero8(), acc1 = vzero8(), acc2 = vzero8(), acc3 = vzero8();

  const int ar  = tid >> 2;        // A row 0..63
  const int ac  = (tid & 3) * 8;   // A col 0,8,16,24
  const int wkk = tid >> 3;        // W k-row 0..31
  const int wnn = (tid & 7) * 16;  // W n-col 0,16,..,112

  for (int k0 = 0; k0 < K; k0 += 32) {
    // ---- stage A tile 64x32 into LDS (convert f32->bf16 if needed) ----
    if (A_IS_BF16) {
      const bf16* A = (const bf16*)Av;
      *(v8bf*)&As[ar * 48 + ac] =
          *(const v8bf*)(A + (size_t)(m0 + ar) * K + k0 + ac);
    } else {
      const float* A = (const float*)Av;
      v8f f = *(const v8f*)(A + (size_t)(m0 + ar) * K + k0 + ac);
      v8bf u;
#pragma unroll
      for (int j = 0; j < 8; j++) u[j] = (bf16)f[j];
      *(v8bf*)&As[ar * 48 + ac] = u;
    }
    // ---- stage W tile 32x128 transposed into LDS as [n][k] ----
    {
      const float* wp = W + (size_t)(k0 + wkk) * N + n0 + wnn;
      v8f f0 = *(const v8f*)(wp);
      v8f f1 = *(const v8f*)(wp + 8);
#pragma unroll
      for (int j = 0; j < 8; j++) Ws[(wnn + j) * 48 + wkk] = (bf16)f0[j];
#pragma unroll
      for (int j = 0; j < 8; j++) Ws[(wnn + 8 + j) * 48 + wkk] = (bf16)f1[j];
    }
    __syncthreads();

    // ---- fragments + WMMA: one A frag feeds 4 B frags ----
    v16bf a;
    {
      const bf16* p = &As[(wm * 16 + ln) * 48 + (hl ? 8 : 0)];
      ((v8bf*)&a)[0] = *(const v8bf*)(p);       // K 0..7   (or 8..15)
      ((v8bf*)&a)[1] = *(const v8bf*)(p + 16);  // K 16..23 (or 24..31)
    }
#pragma unroll
    for (int ns = 0; ns < 4; ns++) {
      v16bf bfr;
      const bf16* p = &Ws[(wn * 64 + ns * 16 + ln) * 48 + (hl ? 8 : 0)];
      ((v8bf*)&bfr)[0] = *(const v8bf*)(p);      // K 0..7   (or 8..15)
      ((v8bf*)&bfr)[1] = *(const v8bf*)(p + 16); // K 16..23 (or 24..31)
      if (ns == 0)      acc0 = wmma_bf16(a, bfr, acc0);
      else if (ns == 1) acc1 = wmma_bf16(a, bfr, acc1);
      else if (ns == 2) acc2 = wmma_bf16(a, bfr, acc2);
      else              acc3 = wmma_bf16(a, bfr, acc3);
    }
    __syncthreads();
  }

  // ---- epilogue: bias + f32 store ----
#pragma unroll
  for (int ns = 0; ns < 4; ns++) {
    v8f acc = (ns == 0) ? acc0 : (ns == 1) ? acc1 : (ns == 2) ? acc2 : acc3;
    int col = n0 + wn * 64 + ns * 16 + ln;
    float bv = bias[col];
#pragma unroll
    for (int r = 0; r < 8; r++) {
      int row = m0 + wm * 16 + r + 8 * hl;
      C[(size_t)row * N + col] = acc[r] + bv;
    }
  }
}

// ---------------------------------------------------------------------------
// RoPE (full d_model, matching reference) + head split, f32 -> bf16 [B,H,T,DT]
// ---------------------------------------------------------------------------
__global__ void __launch_bounds__(256)
rope_split_kernel(const float* __restrict__ qp, const float* __restrict__ kp,
                  const float* __restrict__ vp, bf16* __restrict__ Qh,
                  bf16* __restrict__ Kh, bf16* __restrict__ Vh) {
  size_t i = (size_t)blockIdx.x * 256 + threadIdx.x;  // over B*T*D1
  int d    = (int)(i & (D1_ - 1));
  size_t bt = i >> 10;
  int t    = (int)(bt & (T_ - 1));
  int b    = (int)(bt >> 11);

  int j = d & 511;  // freq index (D1/2 = 512)
  float inv_freq = __expf(-(float)j * (9.210340371976184f / 512.f));  // 10000^(-j/512)
  float ang = (float)t * inv_freq;
  float cs = cosf(ang);
  float sn = sinf(ang);

  float qv = qp[i];
  float kv = kp[i];
  float qr = (d < 512) ? -qp[i + 512] : qp[i - 512];
  float kr = (d < 512) ? -kp[i + 512] : kp[i - 512];
  float qo = qv * cs + qr * sn;
  float ko = kv * cs + kr * sn;

  int h = d >> 6, dt = d & 63;
  size_t o = (((size_t)b * H_ + h) * T_ + t) * DT_ + dt;
  Qh[o] = (bf16)qo;
  Kh[o] = (bf16)ko;
  Vh[o] = (bf16)vp[i];
}

// ---------------------------------------------------------------------------
// Flash attention: one (b,h) and 128 q-rows per block; 8 waves x 16 q-rows.
// Online softmax in registers; P converted C-layout->A-layout via per-wave LDS.
// ---------------------------------------------------------------------------
__global__ void __launch_bounds__(256)
attn_kernel(const bf16* __restrict__ Qh, const bf16* __restrict__ Kh,
            const bf16* __restrict__ Vh, const float* __restrict__ mask,
            bf16* __restrict__ Oa) {
  const int bh = blockIdx.x;
  const int b  = bh >> 4;
  const int h  = bh & 15;
  const int q0 = blockIdx.y * 128;
  const int tid  = threadIdx.x;
  const int lane = tid & 31;
  const int w    = tid >> 5;
  const int ln   = lane & 15;
  const int hl   = lane >> 4;

  const bf16* Qb = Qh + (size_t)bh * T_ * DT_;
  const bf16* Kb = Kh + (size_t)bh * T_ * DT_;
  const bf16* Vb = Vh + (size_t)bh * T_ * DT_;
  const float* Mb = mask + (size_t)b * T_ * T_;

  __shared__ __align__(16) bf16 Ks[128 * 64];   // [kk][dt]  16KB
  __shared__ __align__(16) bf16 Vs[64 * 128];   // [dt][kk]  16KB (transposed)
  __shared__ __align__(16) bf16 Pb[8][16 * 32]; // per-wave P staging, 8KB

  // Q fragments: 16 rows x 64 (two K=32 steps), loaded once from global
  v16bf qa0, qa1;
  {
    const bf16* qp = Qb + (size_t)(q0 + w * 16 + ln) * DT_;
    ((v8bf*)&qa0)[0] = *(const v8bf*)(qp + (hl ? 8 : 0));
    ((v8bf*)&qa0)[1] = *(const v8bf*)(qp + 16 + (hl ? 8 : 0));
    ((v8bf*)&qa1)[0] = *(const v8bf*)(qp + 32 + (hl ? 8 : 0));
    ((v8bf*)&qa1)[1] = *(const v8bf*)(qp + 48 + (hl ? 8 : 0));
  }

  float m[8], l[8];
  v8f o0 = vzero8(), o1 = vzero8(), o2 = vzero8(), o3 = vzero8();
#pragma unroll
  for (int r = 0; r < 8; r++) { m[r] = -1e30f; l[r] = 0.f; }

  const float LOG2E = 1.44269504f;

  for (int k0 = 0; k0 < T_; k0 += 128) {
    // ---- stage K tile (row-major) and V tile (transposed) ----
#pragma unroll
    for (int it = 0; it < 4; it++) {
      int e = (it * 256 + tid) * 8;
      int r = e >> 6, c = e & 63;
      *(v8bf*)&Ks[e] = *(const v8bf*)(Kb + (size_t)(k0 + r) * DT_ + c);
    }
#pragma unroll
    for (int it = 0; it < 4; it++) {
      int e = (it * 256 + tid) * 8;
      int r = e >> 6, c = e & 63;
      v8bf tv = *(const v8bf*)(Vb + (size_t)(k0 + r) * DT_ + c);
#pragma unroll
      for (int j = 0; j < 8; j++) Vs[(c + j) * 128 + r] = tv[j];
    }
    if (k0 + 128 < T_) {
      __builtin_prefetch(Kb + (size_t)(k0 + 128) * DT_, 0, 1);
      __builtin_prefetch(Vb + (size_t)(k0 + 128) * DT_, 0, 1);
    }
    __syncthreads();

    // ---- S = Q @ K^T  (16 q-rows x 128 kv-cols per wave) ----
    v8f sc[8];
#pragma unroll
    for (int nc = 0; nc < 8; nc++) sc[nc] = vzero8();
#pragma unroll
    for (int nc = 0; nc < 8; nc++) {
      v16bf kb0, kb1;
      const bf16* p0 = &Ks[(nc * 16 + ln) * 64 + (hl ? 8 : 0)];
      ((v8bf*)&kb0)[0] = *(const v8bf*)(p0);
      ((v8bf*)&kb0)[1] = *(const v8bf*)(p0 + 16);
      const bf16* p1 = p0 + 32;
      ((v8bf*)&kb1)[0] = *(const v8bf*)(p1);
      ((v8bf*)&kb1)[1] = *(const v8bf*)(p1 + 16);
      sc[nc] = wmma_bf16(qa0, kb0, sc[nc]);
      sc[nc] = wmma_bf16(qa1, kb1, sc[nc]);
    }

    // ---- scale + multiplicative mask ----
#pragma unroll
    for (int nc = 0; nc < 8; nc++) {
      int col = k0 + nc * 16 + ln;
#pragma unroll
      for (int r = 0; r < 8; r++) {
        int row = q0 + w * 16 + r + 8 * hl;
        sc[nc][r] = sc[nc][r] * 0.125f * Mb[(size_t)row * T_ + col];
      }
    }

    // ---- online softmax: row max / exp / row sum (16-lane reductions) ----
    float alpha[8];
#pragma unroll
    for (int r = 0; r < 8; r++) {
      float cm = sc[0][r];
#pragma unroll
      for (int nc = 1; nc < 8; nc++) cm = fmaxf(cm, sc[nc][r]);
#pragma unroll
      for (int off = 1; off < 16; off <<= 1)
        cm = fmaxf(cm, __shfl_xor(cm, off, 32));
      float mn = fmaxf(m[r], cm);
      alpha[r] = __builtin_exp2f((m[r] - mn) * LOG2E);
      m[r] = mn;
    }
    float rs[8];
#pragma unroll
    for (int r = 0; r < 8; r++) rs[r] = 0.f;
#pragma unroll
    for (int nc = 0; nc < 8; nc++)
#pragma unroll
      for (int r = 0; r < 8; r++) {
        float p = __builtin_exp2f((sc[nc][r] - m[r]) * LOG2E);
        sc[nc][r] = p;
        rs[r] += p;
      }
#pragma unroll
    for (int r = 0; r < 8; r++) {
      float s = rs[r];
#pragma unroll
      for (int off = 1; off < 16; off <<= 1) s += __shfl_xor(s, off, 32);
      l[r] = l[r] * alpha[r] + s;
    }
#pragma unroll
    for (int r = 0; r < 8; r++) {
      o0[r] *= alpha[r]; o1[r] *= alpha[r];
      o2[r] *= alpha[r]; o3[r] *= alpha[r];
    }

    // ---- O += P @ V : per 32-wide k-step, stage P slice, same-wave LDS ----
#pragma unroll
    for (int s = 0; s < 4; s++) {
#pragma unroll
      for (int q = 0; q < 2; q++) {
        int nc = 2 * s + q;
#pragma unroll
        for (int r = 0; r < 8; r++)
          Pb[w][(r + 8 * hl) * 32 + q * 16 + ln] = (bf16)sc[nc][r];
      }
      v16bf pa;
      const bf16* pp = &Pb[w][ln * 32 + (hl ? 8 : 0)];
      ((v8bf*)&pa)[0] = *(const v8bf*)(pp);
      ((v8bf*)&pa)[1] = *(const v8bf*)(pp + 16);
#pragma unroll
      for (int ns = 0; ns < 4; ns++) {
        v16bf vf;
        const bf16* vq = &Vs[(ns * 16 + ln) * 128 + s * 32 + (hl ? 8 : 0)];
        ((v8bf*)&vf)[0] = *(const v8bf*)(vq);
        ((v8bf*)&vf)[1] = *(const v8bf*)(vq + 16);
        if (ns == 0) o0 = wmma_bf16(pa, vf, o0);
        else if (ns == 1) o1 = wmma_bf16(pa, vf, o1);
        else if (ns == 2) o2 = wmma_bf16(pa, vf, o2);
        else o3 = wmma_bf16(pa, vf, o3);
      }
    }
    __syncthreads();
  }

  // ---- finalize: divide by row sum, write bf16 [B,T,D1] ----
#pragma unroll
  for (int r = 0; r < 8; r++) {
    float inv = 1.f / l[r];
    int row = q0 + w * 16 + r + 8 * hl;
    size_t base = ((size_t)b * T_ + row) * D1_ + h * DT_;
    Oa[base + 0 * 16 + ln] = (bf16)(o0[r] * inv);
    Oa[base + 1 * 16 + ln] = (bf16)(o1[r] * inv);
    Oa[base + 2 * 16 + ln] = (bf16)(o2[r] * inv);
    Oa[base + 3 * 16 + ln] = (bf16)(o3[r] * inv);
  }
}

// ---------------------------------------------------------------------------
extern "C" void kernel_launch(void* const* d_in, const int* in_sizes, int n_in,
                              void* d_out, int out_size, void* d_ws, size_t ws_size,
                              hipStream_t stream) {
  const float* q    = (const float*)d_in[0];
  const float* k    = (const float*)d_in[1];
  const float* v    = (const float*)d_in[2];
  const float* mask = (const float*)d_in[3];
  const float* wq   = (const float*)d_in[4];
  const float* bq   = (const float*)d_in[5];
  const float* wk   = (const float*)d_in[6];
  const float* bk   = (const float*)d_in[7];
  const float* wv   = (const float*)d_in[8];
  const float* bv   = (const float*)d_in[9];
  const float* wo   = (const float*)d_in[10];
  const float* bo   = (const float*)d_in[11];
  float* out = (float*)d_out;

  const size_t MD = (size_t)B_ * T_ * D1_;  // 8M elements
  char* ws = (char*)d_ws;
  float* qp = (float*)ws; ws += MD * 4;
  float* kp = (float*)ws; ws += MD * 4;
  float* vp = (float*)ws; ws += MD * 4;
  bf16* Qh = (bf16*)ws;  ws += MD * 2;
  bf16* Kh = (bf16*)ws;  ws += MD * 2;
  bf16* Vh = (bf16*)ws;  ws += MD * 2;
  bf16* At = (bf16*)ws;  ws += MD * 2;

  dim3 blk(256);
  dim3 gg(D1_ / 128, (B_ * T_) / 64);

  gemm_kernel<0><<<gg, blk, 0, stream>>>((const void*)q, wq, bq, qp, B_ * T_, D1_, D1_);
  gemm_kernel<0><<<gg, blk, 0, stream>>>((const void*)k, wk, bk, kp, B_ * T_, D1_, D2_);
  gemm_kernel<0><<<gg, blk, 0, stream>>>((const void*)v, wv, bv, vp, B_ * T_, D1_, D2_);

  rope_split_kernel<<<dim3((unsigned)(MD / 256)), blk, 0, stream>>>(qp, kp, vp, Qh, Kh, Vh);

  attn_kernel<<<dim3(B_ * H_, T_ / 128), blk, 0, stream>>>(Qh, Kh, Vh, mask, At);

  gemm_kernel<1><<<gg, blk, 0, stream>>>((const void*)At, wo, bo, out, B_ * T_, D1_, D1_);
}
